// PCDAlignment_4209067950280
// MI455X (gfx1250) — matrix operand order repaired and checked by reading, not compiled, not run on hardware
//
#include <hip/hip_runtime.h>
#include <hip/hip_bf16.h>
#include <math.h>

typedef __attribute__((ext_vector_type(16))) _Float16 h16;
typedef __attribute__((ext_vector_type(8)))  _Float16 h8;
typedef __attribute__((ext_vector_type(8)))  float    f8;

#define BLOCK_THREADS 128
#define WAVES_PER_BLOCK 4
#define PIX_TILE 32            // pixels per block (2 WMMA column groups)
#define KCH 576                // K-chunk staged in LDS (multiple of 32)
#define LDS_STRIDE (KCH + 8)   // +8 halves: bank skew, keeps 16B alignment
#define MAX_OCT_PER_WAVE 4     // supports CoutPad up to 16*4*4 = 256

// Wave index as a *scalar* (SGPR): wave-uniform by construction; readfirstlane
// makes that explicit so oc-tile guards compile to s_cmp/s_cbranch (EXEC
// untouched) instead of divergent saveexec blocks.  WMMA requires EXEC ==
// all-ones, so this is a correctness requirement, not just perf.
__device__ __forceinline__ int wave_id() {
  return (int)(__builtin_amdgcn_readfirstlane((unsigned)threadIdx.x) >> 5);
}

// ---------------------------------------------------------------------------
// Weight prep: repack w[Cout][Ctot][3][3] fp32 -> f16 wbuf[CoutPad][Kpad],
// K index = ci*9 + (kh*3+kw); bias padded to CoutPad fp32.
// ---------------------------------------------------------------------------
__global__ void prep_weights_kernel(const float* __restrict__ w,
                                    const float* __restrict__ b,
                                    int Cout, int Ctot,
                                    _Float16* __restrict__ wbuf,
                                    float* __restrict__ bbuf,
                                    int CoutPad, int Kpad) {
  long idx = (long)blockIdx.x * blockDim.x + threadIdx.x;
  int K = Ctot * 9;
  long total = (long)CoutPad * Kpad;
  if (idx < total) {
    int oc = (int)(idx / Kpad);
    int k  = (int)(idx - (long)oc * Kpad);
    float v = 0.f;
    if (oc < Cout && k < K) v = w[(long)oc * K + k];
    wbuf[idx] = (_Float16)v;
  }
  if (idx < CoutPad) bbuf[idx] = (idx < Cout) ? b[idx] : 0.f;
}

// ---------------------------------------------------------------------------
// One K-chunk of the WMMA GEMM (KPAD = compile-time A row stride so address
// math constant-folds into load immediates). Each wave owns up to 4 oc-tiles
// x 2 pixel groups of accumulators. B fragments (LDS) are loaded once per
// k-step and reused across oc-tiles; A fragments (global f16 weights,
// L2-resident) are reused across the 2 pixel groups.
// A 16x32 f16 layout: lane<16 -> K {0..7,16..23}, lane>=16 -> K {8..15,24..31},
// M = lane&15.  C/D layout: M = r + 8*(lane>>4), N = lane&15.
// ---------------------------------------------------------------------------
template <int KPAD>
__device__ __forceinline__ void wmma_chunk_accum(
    const _Float16* __restrict__ Blds, const _Float16* __restrict__ wbuf,
    int kc, f8 (&acc)[MAX_OCT_PER_WAVE][2], int nOcT) {
  int wave = wave_id();                 // SGPR
  int lane = threadIdx.x & 31;
  int hi   = lane >> 4;
  int mrow = lane & 15;
  int ahalf = hi << 3;   // 0 or 8
  int col  = lane & 15;
  int bhalf = hi << 3;   // 0 or 8
  const _Float16* bp0 = Blds + (long)col * LDS_STRIDE + bhalf;
  const _Float16* bp1 = Blds + (long)(16 + col) * LDS_STRIDE + bhalf;
  // Per-tile A base pointers: KPAD is a constant, so mrow*KPAD is a single
  // constant-mul hoisted out of the k-loop.
  const _Float16* ap[MAX_OCT_PER_WAVE];
#pragma unroll
  for (int t = 0; t < MAX_OCT_PER_WAVE; ++t) {
    int ot = wave + (t << 2);
    ap[t] = wbuf + (long)(((ot < nOcT ? ot : 0) << 4) + mrow) * KPAD + kc + ahalf;
  }
  for (int kt = 0; kt < KCH; kt += 32) {
    h8 b0a = *(const h8*)(bp0 + kt);
    h8 b0b = *(const h8*)(bp0 + kt + 16);
    h16 B0 = __builtin_shufflevector(b0a, b0b, 0, 1, 2, 3, 4, 5, 6, 7,
                                     8, 9, 10, 11, 12, 13, 14, 15);
    h8 b1a = *(const h8*)(bp1 + kt);
    h8 b1b = *(const h8*)(bp1 + kt + 16);
    h16 B1 = __builtin_shufflevector(b1a, b1b, 0, 1, 2, 3, 4, 5, 6, 7,
                                     8, 9, 10, 11, 12, 13, 14, 15);
#pragma unroll
    for (int t = 0; t < MAX_OCT_PER_WAVE; ++t) {
      int ot = wave + (t << 2);         // scalar
      if (ot < nOcT) {                  // scalar branch: EXEC stays all-ones
        h8 a0 = *(const h8*)(ap[t] + kt);
        h8 a1 = *(const h8*)(ap[t] + kt + 16);
        h16 A = __builtin_shufflevector(a0, a1, 0, 1, 2, 3, 4, 5, 6, 7,
                                        8, 9, 10, 11, 12, 13, 14, 15);
        acc[t][0] = __builtin_amdgcn_wmma_f32_16x16x32_f16(
            false, A, false, B0, (short)0, acc[t][0], false, false);
        acc[t][1] = __builtin_amdgcn_wmma_f32_16x16x32_f16(
            false, A, false, B1, (short)0, acc[t][1], false, false);
      }
    }
  }
}

__device__ __forceinline__ void acc_init(f8 (&acc)[MAX_OCT_PER_WAVE][2],
                                         const float* __restrict__ bbuf,
                                         int nOcT) {
  int wave = wave_id();
  int hi = (threadIdx.x & 31) >> 4;
#pragma unroll
  for (int t = 0; t < MAX_OCT_PER_WAVE; ++t) {
    int ot = wave + (t << 2);
    if (ot < nOcT) {
#pragma unroll
      for (int r = 0; r < 8; ++r) {
        float bv = bbuf[(ot << 4) + (hi << 3) + r];
        acc[t][0][r] = bv;
        acc[t][1][r] = bv;
      }
    }
  }
}

__device__ __forceinline__ void acc_store(const f8 (&acc)[MAX_OCT_PER_WAVE][2],
                                          float* __restrict__ out, int b, int y,
                                          int px0, int Hh, int Ww, int Cout,
                                          int nOcT, int lrelu) {
  int wave = wave_id();
  int lane = threadIdx.x & 31;
  int hi = lane >> 4;
  int col = lane & 15;
#pragma unroll
  for (int t = 0; t < MAX_OCT_PER_WAVE; ++t) {
    int ot = wave + (t << 2);
    if (ot < nOcT) {
#pragma unroll
      for (int g = 0; g < 2; ++g) {
#pragma unroll
        for (int r = 0; r < 8; ++r) {
          int m = (ot << 4) + (hi << 3) + r;
          if (m < Cout) {
            float v = acc[t][g][r];
            if (lrelu) v = v > 0.f ? v : 0.1f * v;
            out[(((long)b * Cout + m) * Hh + y) * Ww + px0 + (g << 4) + col] = v;
          }
        }
      }
    }
  }
}

// ---------------------------------------------------------------------------
// Implicit-GEMM 3x3 conv, stride 1, pad 1, dual-input channel concat.
// Block = 4 waves: 32 pixels x CoutPad outputs; K staged in LDS in 576-chunks.
// KPAD in {576, 1152} (== Ctot*9 for Ctot in {64, 128}).
// ---------------------------------------------------------------------------
template <int KPAD>
__global__ __launch_bounds__(BLOCK_THREADS) void conv3x3_wmma_kernel(
    const float* __restrict__ x1, int c1, const float* __restrict__ x2, int c2,
    const _Float16* __restrict__ wbuf, const float* __restrict__ bbuf,
    float* __restrict__ out, int Bn, int Hh, int Ww,
    int Cout, int CoutPad, int lrelu) {
  __shared__ _Float16 Blds[PIX_TILE * LDS_STRIDE];
  int Ctot = c1 + c2;
  int K = Ctot * 9;  // == KPAD for all instantiated cases
  int wtiles = Ww >> 5;
  int t = blockIdx.x;
  int px0 = (t % wtiles) << 5;
  int y   = (t / wtiles) % Hh;
  int b   = t / (wtiles * Hh);
  int tid = threadIdx.x;
  int nOcT = CoutPad >> 4;

  f8 acc[MAX_OCT_PER_WAVE][2];
  acc_init(acc, bbuf, nOcT);

  for (int kc = 0; kc < KPAD; kc += KCH) {
    int Kcur = K - kc;
    if (Kcur > KCH) Kcur = KCH;
    for (int k = Kcur + tid; k < KCH; k += BLOCK_THREADS)
      for (int p = 0; p < PIX_TILE; ++p)
        Blds[p * LDS_STRIDE + k] = (_Float16)0.f;
    for (int idx = tid; idx < Kcur; idx += BLOCK_THREADS) {
      int kg = kc + idx;
      int ci = kg / 9, tap = kg - ci * 9;
      int dh = tap / 3 - 1, dw = tap % 3 - 1;
      const float* src = (ci < c1)
          ? (x1 + (long)(b * c1 + ci) * Hh * Ww)
          : (x2 + (long)(b * c2 + (ci - c1)) * Hh * Ww);
      int yy = y + dh;
      bool yok = ((unsigned)yy < (unsigned)Hh);
      const float* row = src + (long)yy * Ww;
#pragma unroll
      for (int p = 0; p < PIX_TILE; ++p) {
        int xx = px0 + p + dw;
        float v = (yok && (unsigned)xx < (unsigned)Ww) ? row[xx] : 0.f;
        Blds[p * LDS_STRIDE + idx] = (_Float16)v;
      }
    }
    __syncthreads();
    wmma_chunk_accum<KPAD>(Blds, wbuf, kc, acc, nOcT);
    __syncthreads();
  }
  acc_store(acc, out, b, y, px0, Hh, Ww, Cout, nOcT, lrelu);
}

// ---------------------------------------------------------------------------
// Modulated deformable conv (DCNv2): bilinear gather x sigmoid(mask) into the
// im2col LDS tile, then the same WMMA GEMM (K = C*9 = 576). off layout:
// [B][3*G*9][H][W]; channels [0,G*9)=off_y, [G*9,2G*9)=off_x, rest=mask.
// ---------------------------------------------------------------------------
__global__ __launch_bounds__(BLOCK_THREADS) void dcn_wmma_kernel(
    const float* __restrict__ x, const float* __restrict__ off,
    const _Float16* __restrict__ wbuf, const float* __restrict__ bbuf,
    float* __restrict__ out, int Bn, int C, int G, int Hh, int Ww,
    int Cout, int CoutPad, int lrelu) {
  __shared__ _Float16 Blds[PIX_TILE * LDS_STRIDE];
  int K = C * 9;  // 576
  int Cg = C / G;
  int GK = G * 9;
  int wtiles = Ww >> 5;
  int t = blockIdx.x;
  int px0 = (t % wtiles) << 5;
  int y   = (t / wtiles) % Hh;
  int b   = t / (wtiles * Hh);
  int tid = threadIdx.x;
  int nOcT = CoutPad >> 4;
  long planeHW = (long)Hh * Ww;

  f8 acc[MAX_OCT_PER_WAVE][2];
  acc_init(acc, bbuf, nOcT);

  for (int kc = 0; kc < KCH; kc += KCH) {  // single 576 chunk
    int Kcur = K - kc;
    if (Kcur > KCH) Kcur = KCH;
    for (int k = Kcur + tid; k < KCH; k += BLOCK_THREADS)
      for (int p = 0; p < PIX_TILE; ++p)
        Blds[p * LDS_STRIDE + k] = (_Float16)0.f;
    for (int idx = tid; idx < Kcur; idx += BLOCK_THREADS) {
      int kg = kc + idx;
      int ci = kg / 9, tap = kg - ci * 9;
      int dh = tap / 3 - 1, dw = tap % 3 - 1;
      int g = ci / Cg;
      const float* oy_p =
          off + ((long)b * (3 * GK) + (g * 9 + tap)) * planeHW + (long)y * Ww;
      const float* ox_p = oy_p + (long)GK * planeHW;
      const float* mm_p = oy_p + 2L * GK * planeHW;
      const float* ch = x + (long)(b * C + ci) * planeHW;
      for (int p = 0; p < PIX_TILE; ++p) {
        int xx = px0 + p;
        float oy = oy_p[xx], ox = ox_p[xx];
        float mm = 1.f / (1.f + __expf(-mm_p[xx]));
        float py = (float)(y + dh) + oy;
        float px = (float)(xx + dw) + ox;
        float y0f = floorf(py), x0f = floorf(px);
        float ty = py - y0f, tx = px - x0f;
        int y0 = (int)y0f, x0 = (int)x0f;
        float v = 0.f;
#pragma unroll
        for (int j = 0; j < 2; ++j) {
          int iy = y0 + j;
          float wy = j ? ty : 1.f - ty;
          if ((unsigned)iy < (unsigned)Hh) {
#pragma unroll
            for (int i2 = 0; i2 < 2; ++i2) {
              int ix = x0 + i2;
              float wx = i2 ? tx : 1.f - tx;
              if ((unsigned)ix < (unsigned)Ww)
                v += wy * wx * ch[(long)iy * Ww + ix];
            }
          }
        }
        Blds[p * LDS_STRIDE + idx] = (_Float16)(v * mm);
      }
    }
    __syncthreads();
    wmma_chunk_accum<KCH>(Blds, wbuf, kc, acc, nOcT);
    __syncthreads();
  }
  acc_store(acc, out, b, y, px0, Hh, Ww, Cout, nOcT, lrelu);
}

// ---------------------------------------------------------------------------
// Bilinear 2x upsample (half-pixel centers, edge clamp), fused output scale.
// ---------------------------------------------------------------------------
__global__ void upsample2x_kernel(const float* __restrict__ in,
                                  float* __restrict__ out,
                                  int Bn, int C, int Hin, int Win, float scale) {
  long i = (long)blockIdx.x * blockDim.x + threadIdx.x;
  int Ho = Hin * 2, Wo = Win * 2;
  long total = (long)Bn * C * Ho * Wo;
  if (i >= total) return;
  int xo = (int)(i % Wo);
  int yo = (int)((i / Wo) % Ho);
  long bc = i / ((long)Wo * Ho);
  float sy = (yo + 0.5f) * 0.5f - 0.5f;
  float sx = (xo + 0.5f) * 0.5f - 0.5f;
  float y0f = floorf(sy), x0f = floorf(sx);
  float ty = sy - y0f, tx = sx - x0f;
  int y0 = (int)y0f, x0 = (int)x0f;
  int y1 = y0 + 1, x1 = x0 + 1;
  y0 = y0 < 0 ? 0 : (y0 >= Hin ? Hin - 1 : y0);
  y1 = y1 < 0 ? 0 : (y1 >= Hin ? Hin - 1 : y1);
  x0 = x0 < 0 ? 0 : (x0 >= Win ? Win - 1 : x0);
  x1 = x1 < 0 ? 0 : (x1 >= Win ? Win - 1 : x1);
  const float* p = in + bc * Hin * Win;
  float v = (1.f - ty) * ((1.f - tx) * p[(long)y0 * Win + x0] +
                          tx * p[(long)y0 * Win + x1]) +
            ty * ((1.f - tx) * p[(long)y1 * Win + x0] +
                  tx * p[(long)y1 * Win + x1]);
  out[i] = v * scale;
}

// ---------------------------------------------------------------------------
// Host-side helpers
// ---------------------------------------------------------------------------
static void run_conv(hipStream_t s, const float* x1, int c1, const float* x2,
                     int c2, const float* w, const float* b, int Cout,
                     float* out, int Bn, int Hh, int Ww, int lrelu,
                     _Float16* wbuf, float* bbuf) {
  int Ctot = c1 + c2;
  int K = Ctot * 9;
  int Kpad = ((K + KCH - 1) / KCH) * KCH;
  int CoutPad = ((Cout + 15) / 16) * 16;
  long tot = (long)CoutPad * Kpad;
  prep_weights_kernel<<<(unsigned)((tot + 255) / 256), 256, 0, s>>>(
      w, b, Cout, Ctot, wbuf, bbuf, CoutPad, Kpad);
  dim3 grid((unsigned)(Bn * Hh * (Ww / PIX_TILE)));
  if (Kpad == 576) {
    conv3x3_wmma_kernel<576><<<grid, BLOCK_THREADS, 0, s>>>(
        x1, c1, x2, c2, wbuf, bbuf, out, Bn, Hh, Ww, Cout, CoutPad, lrelu);
  } else if (Kpad == 1152) {
    conv3x3_wmma_kernel<1152><<<grid, BLOCK_THREADS, 0, s>>>(
        x1, c1, x2, c2, wbuf, bbuf, out, Bn, Hh, Ww, Cout, CoutPad, lrelu);
  }
}

static void run_dcn(hipStream_t s, const float* x, const float* off,
                    const float* w, const float* b, float* out, int Bn, int C,
                    int G, int Hh, int Ww, int lrelu, _Float16* wbuf,
                    float* bbuf) {
  int Cout = C;
  int K = C * 9;                       // 576
  int Kpad = ((K + KCH - 1) / KCH) * KCH;
  int CoutPad = ((Cout + 15) / 16) * 16;
  long tot = (long)CoutPad * Kpad;
  prep_weights_kernel<<<(unsigned)((tot + 255) / 256), 256, 0, s>>>(
      w, b, Cout, C, wbuf, bbuf, CoutPad, Kpad);
  dim3 grid((unsigned)(Bn * Hh * (Ww / PIX_TILE)));
  dcn_wmma_kernel<<<grid, BLOCK_THREADS, 0, s>>>(
      x, off, wbuf, bbuf, out, Bn, C, G, Hh, Ww, Cout, CoutPad, lrelu);
}

static void run_up2(hipStream_t s, const float* in, float* out, int Bn, int C,
                    int Hin, int Win, float scale) {
  long total = (long)Bn * C * Hin * 2 * Win * 2;
  upsample2x_kernel<<<(unsigned)((total + 255) / 256), 256, 0, s>>>(
      in, out, Bn, C, Hin, Win, scale);
}

struct LvlP {
  const float *oc1w, *oc1b, *oc2w, *oc2b, *oc3w, *oc3b, *fcw, *fcb;
  const float *dcnb, *dcncob, *dcncow, *dcnw;
};

extern "C" void kernel_launch(void* const* d_in, const int* in_sizes, int n_in,
                              void* d_out, int out_size, void* d_ws,
                              size_t ws_size, hipStream_t stream) {
  if (n_in < 47) return;
  const int Bn = 16, C = 64, G = 8;
  // JAX pytree flatten order (dict keys sorted): n_groups, nbr_l1..l3,
  // params{cas,l1,l2,l3} (each dict sorted: dcn{b,co_b,co_w,w}, fc, oc1(w,b),
  // oc2, oc3), ref_l1..l3.
  const float* nbr1 = (const float*)d_in[1];
  const float* nbr2 = (const float*)d_in[2];
  const float* nbr3 = (const float*)d_in[3];
  const float* ref1 = (const float*)d_in[44];
  const float* ref2 = (const float*)d_in[45];
  const float* ref3 = (const float*)d_in[46];

  LvlP cas, l1, l2, l3;
  cas.dcnb = (const float*)d_in[4];  cas.dcncob = (const float*)d_in[5];
  cas.dcncow = (const float*)d_in[6]; cas.dcnw = (const float*)d_in[7];
  cas.oc1w = (const float*)d_in[8];  cas.oc1b = (const float*)d_in[9];
  cas.oc2w = (const float*)d_in[10]; cas.oc2b = (const float*)d_in[11];
  cas.oc3w = cas.oc3b = cas.fcw = cas.fcb = nullptr;

  LvlP* mid[2] = {&l1, &l2};
  for (int li = 0; li < 2; ++li) {
    int base = 12 + 12 * li;
    LvlP* L = mid[li];
    L->dcnb   = (const float*)d_in[base + 0];
    L->dcncob = (const float*)d_in[base + 1];
    L->dcncow = (const float*)d_in[base + 2];
    L->dcnw   = (const float*)d_in[base + 3];
    L->fcw    = (const float*)d_in[base + 4];
    L->fcb    = (const float*)d_in[base + 5];
    L->oc1w   = (const float*)d_in[base + 6];
    L->oc1b   = (const float*)d_in[base + 7];
    L->oc2w   = (const float*)d_in[base + 8];
    L->oc2b   = (const float*)d_in[base + 9];
    L->oc3w   = (const float*)d_in[base + 10];
    L->oc3b   = (const float*)d_in[base + 11];
  }
  l3.dcnb = (const float*)d_in[36]; l3.dcncob = (const float*)d_in[37];
  l3.dcncow = (const float*)d_in[38]; l3.dcnw = (const float*)d_in[39];
  l3.oc1w = (const float*)d_in[40]; l3.oc1b = (const float*)d_in[41];
  l3.oc2w = (const float*)d_in[42]; l3.oc2b = (const float*)d_in[43];
  l3.oc3w = l3.oc3b = l3.fcw = l3.fcb = nullptr;

  // ---- workspace carve-out ----
  char* wsp = (char*)d_ws;
  size_t cur = 0;
  auto take = [&](size_t bytes) -> char* {
    char* p = wsp + cur;
    cur += (bytes + 255) & ~(size_t)255;
    return p;
  };
  const size_t NL1 = (size_t)Bn * C * 128 * 128;  // 16.78M elems
  float* bufA  = (float*)take(NL1 * 4);
  float* bufB  = (float*)take(NL1 * 4);
  float* OFFb  = (float*)take(NL1 * 4);
  float* FEATb = (float*)take(NL1 * 4);
  float* UPOb  = (float*)take(NL1 * 4);
  float* UPFb  = (float*)take(NL1 * 4);
  float* DCNOb = (float*)take((size_t)Bn * (3 * G * 9) * 128 * 128 * 4);
  _Float16* WBUF = (_Float16*)take((size_t)224 * 1152 * 2);
  float* BBUF = (float*)take(224 * 4);
  if (cur > ws_size) return;  // workspace too small: refuse to scribble

  hipStream_t s = stream;
  const int OffC = 3 * G * 9;  // 216

  // ---------- level 3 (32x32) ----------
  {
    int Hh = 32, Ww = 32;
    run_conv(s, nbr3, C, ref3, C, l3.oc1w, l3.oc1b, C, bufA, Bn, Hh, Ww, 1, WBUF, BBUF);
    run_conv(s, bufA, C, nullptr, 0, l3.oc2w, l3.oc2b, C, OFFb, Bn, Hh, Ww, 1, WBUF, BBUF);
    run_conv(s, OFFb, C, nullptr, 0, l3.dcncow, l3.dcncob, OffC, DCNOb, Bn, Hh, Ww, 0, WBUF, BBUF);
    run_dcn(s, nbr3, DCNOb, l3.dcnw, l3.dcnb, FEATb, Bn, C, G, Hh, Ww, 1, WBUF, BBUF);
    run_up2(s, OFFb, UPOb, Bn, C, Hh, Ww, 2.0f);
    run_up2(s, FEATb, UPFb, Bn, C, Hh, Ww, 1.0f);
  }
  // ---------- level 2 (64x64) ----------
  {
    int Hh = 64, Ww = 64;
    run_conv(s, nbr2, C, ref2, C, l2.oc1w, l2.oc1b, C, bufA, Bn, Hh, Ww, 1, WBUF, BBUF);
    run_conv(s, bufA, C, UPOb, C, l2.oc2w, l2.oc2b, C, bufB, Bn, Hh, Ww, 1, WBUF, BBUF);
    run_conv(s, bufB, C, nullptr, 0, l2.oc3w, l2.oc3b, C, OFFb, Bn, Hh, Ww, 1, WBUF, BBUF);
    run_conv(s, OFFb, C, nullptr, 0, l2.dcncow, l2.dcncob, OffC, DCNOb, Bn, Hh, Ww, 0, WBUF, BBUF);
    run_dcn(s, nbr2, DCNOb, l2.dcnw, l2.dcnb, bufA, Bn, C, G, Hh, Ww, 0, WBUF, BBUF);
    run_conv(s, bufA, C, UPFb, C, l2.fcw, l2.fcb, C, FEATb, Bn, Hh, Ww, 1, WBUF, BBUF);
    run_up2(s, OFFb, UPOb, Bn, C, Hh, Ww, 2.0f);
    run_up2(s, FEATb, UPFb, Bn, C, Hh, Ww, 1.0f);
  }
  // ---------- level 1 (128x128) ----------
  {
    int Hh = 128, Ww = 128;
    run_conv(s, nbr1, C, ref1, C, l1.oc1w, l1.oc1b, C, bufA, Bn, Hh, Ww, 1, WBUF, BBUF);
    run_conv(s, bufA, C, UPOb, C, l1.oc2w, l1.oc2b, C, bufB, Bn, Hh, Ww, 1, WBUF, BBUF);
    run_conv(s, bufB, C, nullptr, 0, l1.oc3w, l1.oc3b, C, OFFb, Bn, Hh, Ww, 1, WBUF, BBUF);
    run_conv(s, OFFb, C, nullptr, 0, l1.dcncow, l1.dcncob, OffC, DCNOb, Bn, Hh, Ww, 0, WBUF, BBUF);
    run_dcn(s, nbr1, DCNOb, l1.dcnw, l1.dcnb, bufA, Bn, C, G, Hh, Ww, 0, WBUF, BBUF);
    run_conv(s, bufA, C, UPFb, C, l1.fcw, l1.fcb, C, FEATb, Bn, Hh, Ww, 0, WBUF, BBUF);
  }
  // ---------- cascade refinement (128x128) ----------
  {
    int Hh = 128, Ww = 128;
    run_conv(s, FEATb, C, ref1, C, cas.oc1w, cas.oc1b, C, bufA, Bn, Hh, Ww, 1, WBUF, BBUF);
    run_conv(s, bufA, C, nullptr, 0, cas.oc2w, cas.oc2b, C, bufB, Bn, Hh, Ww, 1, WBUF, BBUF);
    run_conv(s, bufB, C, nullptr, 0, cas.dcncow, cas.dcncob, OffC, DCNOb, Bn, Hh, Ww, 0, WBUF, BBUF);
    run_dcn(s, FEATb, DCNOb, cas.dcnw, cas.dcnb, (float*)d_out, Bn, C, G, Hh, Ww, 1, WBUF, BBUF);
  }
}
